// Atten_Head_72834055405568
// MI455X (gfx1250) — matrix operand order repaired and checked by reading, not compiled, not run on hardware
//
#include <hip/hip_runtime.h>
#include <hip/hip_bf16.h>

typedef __bf16 bf16;
typedef __attribute__((ext_vector_type(16))) __bf16 v16bf;
typedef __attribute__((ext_vector_type(8)))  __bf16 v8bf;
typedef __attribute__((ext_vector_type(8)))  float  v8f;
typedef __attribute__((ext_vector_type(4)))  float  v4f;

#define D_MODEL 1024
#define SEQ     2048
#define NBATCH  4
#define NTOK    (SEQ * NBATCH)          // 8192
#define QS_LD   1032                    // Q tile LDS row stride (conflict-free)
#define PT_LD   40                      // P tile LDS row stride (80B, 16B aligned)
#define KS_LD   136                     // staged-K LDS row stride (272B: conflict-free b128 reads)
#define KS_WAVE (32 * KS_LD)            // one wave's 32x128 K slice (padded)

__device__ __forceinline__ v8f zero8() {
    v8f z;
#pragma unroll
    for (int i = 0; i < 8; i++) z[i] = 0.f;
    return z;
}

__device__ __forceinline__ v8f wmma_bf16(v16bf a, v16bf b, v8f c) {
    return __builtin_amdgcn_wmma_f32_16x16x32_bf16(
        false, a, false, b, (short)0, c, false, false);
}

// A fragment 16x32 bf16 from row-major bf16 (K = e + 16*half)
__device__ __forceinline__ v16bf load_a_bf16rows(const bf16* base, int ld,
                                                 int row, int k0, int half) {
    const bf16* p = base + (size_t)row * ld + k0;
    v8bf x0 = *(const v8bf*)(p + 16 * half);
    v8bf x1 = *(const v8bf*)(p + 16 * half + 8);
    v16bf a;
#pragma unroll
    for (int i = 0; i < 8; i++) { a[i] = x0[i]; a[8 + i] = x1[i]; }
    return a;
}

// B fragment 32x16 bf16: contiguous along K per lane (K = e + 16*half)
__device__ __forceinline__ v16bf load_b_contig(const bf16* p) {
    return *(const v16bf*)p;  // 32B -> two global_load_b128 (or ds_load_b128 x2)
}

// ---- CDNA5 async copy: 16B per lane, global -> LDS, tracked by ASYNCcnt ----
__device__ __forceinline__ void async_copy_b128(bf16* lds_dst, const bf16* gsrc) {
    unsigned dst = (unsigned)(size_t)lds_dst;   // low 32 bits = LDS offset
    asm volatile("global_load_async_to_lds_b128 %0, %1, off"
                 :: "v"(dst), "v"(gsrc) : "memory");
}
__device__ __forceinline__ void wait_async_le16() {
    asm volatile("s_wait_asynccnt 0x10" ::: "memory");
}
__device__ __forceinline__ void wait_async_0() {
    asm volatile("s_wait_asynccnt 0x0" ::: "memory");
}

// Stage one wave's 32-token x 128-dim K slice into LDS (16 async issues).
__device__ __forceinline__ void stage_k(bf16* dst, const bf16* kb_slice, int lane) {
#pragma unroll
    for (int t = 0; t < 16; t++) {
        int c   = t * 32 + lane;      // 16B chunk index; 16 chunks per 256B row
        int row = c >> 4;
        int col = (c & 15) * 8;       // bf16 elements
        async_copy_b128(dst + row * KS_LD + col,
                        kb_slice + (size_t)row * D_MODEL + col);
    }
}

// ---------------------------------------------------------------------------
// Prep 1: X (fp32) -> Xbf (bf16 row-major), 8 elems/thread
// ---------------------------------------------------------------------------
__global__ __launch_bounds__(256) void convert_x_kernel(const float* __restrict__ X,
                                                        bf16* __restrict__ Xbf) {
    size_t i = ((size_t)blockIdx.x * 256 + threadIdx.x) * 8;
    v4f a = *(const v4f*)(X + i);
    v4f b = *(const v4f*)(X + i + 4);
    v8bf o;
#pragma unroll
    for (int k = 0; k < 4; k++) { o[k] = (bf16)a[k]; o[4 + k] = (bf16)b[k]; }
    *(v8bf*)(Xbf + i) = o;
}

// ---------------------------------------------------------------------------
// Prep 2: W[k][n] (fp32) -> Wt[n][k] (bf16), 32x32 tiles through LDS
// ---------------------------------------------------------------------------
__global__ __launch_bounds__(256) void transpose_w_kernel(
    const float* __restrict__ WK, const float* __restrict__ WQ,
    const float* __restrict__ WV, bf16* __restrict__ WtK,
    bf16* __restrict__ WtQ, bf16* __restrict__ WtV) {
    __shared__ bf16 t[32][33];
    const float* W = (blockIdx.z == 0) ? WK : (blockIdx.z == 1) ? WQ : WV;
    bf16*       Wt = (blockIdx.z == 0) ? WtK : (blockIdx.z == 1) ? WtQ : WtV;
    const int k0 = blockIdx.x * 32, n0 = blockIdx.y * 32;
    const int tx = threadIdx.x & 31, ty = threadIdx.x >> 5;
#pragma unroll
    for (int r = ty; r < 32; r += 8)
        t[r][tx] = (bf16)W[(size_t)(k0 + r) * D_MODEL + n0 + tx];
    __syncthreads();
#pragma unroll
    for (int r = ty; r < 32; r += 8)
        Wt[(size_t)(n0 + r) * D_MODEL + k0 + tx] = t[tx][r];
}

// ---------------------------------------------------------------------------
// Kernel 1: K = Xbf*WK (bf16 [token][dim]); V = Xbf*WV stored transposed
//           (bf16 [dim][token]).  One wave per 16x64 tile, A frag reused x4.
// ---------------------------------------------------------------------------
__global__ __launch_bounds__(32) void gemm_kv_kernel(
    const bf16* __restrict__ Xbf, const bf16* __restrict__ WtK,
    const bf16* __restrict__ WtV, bf16* __restrict__ Kbf, bf16* __restrict__ Vt) {
    const int lane = threadIdx.x & 31;
    const int half = lane >> 4;
    const int ln   = lane & 15;
    const int m0   = blockIdx.x * 16;   // token tile
    const int n0   = blockIdx.y * 64;   // dim tile (4 x 16)
    const bool isV = (blockIdx.z != 0);
    const bf16* Wt = isV ? WtV : WtK;

    v8f acc[4];
#pragma unroll
    for (int t = 0; t < 4; t++) acc[t] = zero8();

    for (int kk = 0; kk < D_MODEL; kk += 32) {
        v16bf a = load_a_bf16rows(Xbf, D_MODEL, m0 + ln, kk, half);
#pragma unroll
        for (int t = 0; t < 4; t++) {
            v16bf b = load_b_contig(Wt + (size_t)(n0 + t * 16 + ln) * D_MODEL
                                       + kk + 16 * half);
            acc[t] = wmma_bf16(a, b, acc[t]);
        }
    }
    if (!isV) {
#pragma unroll
        for (int t = 0; t < 4; t++)
#pragma unroll
            for (int v = 0; v < 8; v++) {
                int r = m0 + v + 8 * half;
                Kbf[(size_t)r * D_MODEL + n0 + t * 16 + ln] = (bf16)acc[t][v];
            }
    } else {
#pragma unroll
        for (int t = 0; t < 4; t++) {
            v8bf pk;
#pragma unroll
            for (int v = 0; v < 8; v++) pk[v] = (bf16)acc[t][v];
            *(v8bf*)(Vt + (size_t)(n0 + t * 16 + ln) * NTOK + m0 + 8 * half) = pk;
        }
    }
}

// ---------------------------------------------------------------------------
// Kernel 2: fused Q-projection + causal flash attention.
//   Block = 256 thr (8 waves) per (batch, 16-query tile).
//   Wave w owns D-slice [w*128, +128). K tiles double-buffered in LDS via
//   global_load_async_to_lds_b128 / s_wait_asynccnt.
// ---------------------------------------------------------------------------
__global__ __launch_bounds__(256) void attn_kernel(
    const bf16* __restrict__ Xbf, const bf16* __restrict__ WtQ,
    const bf16* __restrict__ Kbf, const bf16* __restrict__ Vt,
    float* __restrict__ Out) {
    __shared__ bf16  Qs[16 * QS_LD];        // 16x1024 bf16 Q tile (pre-scaled)
    __shared__ bf16  Ks[2][8 * KS_WAVE];    // double-buffered per-wave K slices
    __shared__ float sc[16][32];            // cross-wave score reduction
    __shared__ bf16  Pt[16 * PT_LD];        // softmax probabilities
    __shared__ float Mrow[16], Lrow[16], Arow[16];

    const int tid  = threadIdx.x;
    const int w    = tid >> 5;
    const int lane = tid & 31;
    const int half = lane >> 4;
    const int ln   = lane & 15;
    const int q0   = blockIdx.x * 16;
    const int tokb = blockIdx.y * SEQ;
    const int jmax = (q0 + 15) >> 5;

    bf16* ksw0 = &Ks[0][w * KS_WAVE];
    bf16* ksw1 = &Ks[1][w * KS_WAVE];
    const bf16* kslice = Kbf + (size_t)tokb * D_MODEL + w * 128;

    // Kick off async stage of K tile j=0 behind the Q projection
    stage_k(ksw0, kslice, lane);

    // ---- Phase 1: Q tile projection into LDS, scaled by 1/sqrt(D) ----
    {
        const bf16* Xq = Xbf + (size_t)(tokb + q0) * D_MODEL;
        v8f qacc[8];
#pragma unroll
        for (int st = 0; st < 8; st++) qacc[st] = zero8();
        for (int kk = 0; kk < D_MODEL; kk += 32) {
            v16bf a = load_a_bf16rows(Xq, D_MODEL, ln, kk, half);
#pragma unroll
            for (int st = 0; st < 8; st++) {
                v16bf b = load_b_contig(WtQ + (size_t)(w * 128 + st * 16 + ln) * D_MODEL
                                            + kk + 16 * half);
                qacc[st] = wmma_bf16(a, b, qacc[st]);
            }
        }
        const float scl = 0.03125f;  // 1/sqrt(1024)
#pragma unroll
        for (int st = 0; st < 8; st++)
#pragma unroll
            for (int v = 0; v < 8; v++)
                Qs[(v + 8 * half) * QS_LD + w * 128 + st * 16 + ln] =
                    (bf16)(qacc[st][v] * scl);
    }
    if (tid < 16) { Mrow[tid] = -3.0e38f; Lrow[tid] = 0.f; }
    __syncthreads();

    v8f oacc[8];
#pragma unroll
    for (int st = 0; st < 8; st++) oacc[st] = zero8();

    for (int j = 0; j <= jmax; ++j) {
        const int kv = j * 32;
        bf16* kcur = (j & 1) ? ksw1 : ksw0;
        if (j < jmax) {   // stage next tile, then wait for current (in-order)
            stage_k((j & 1) ? ksw0 : ksw1, kslice + (size_t)(kv + 32) * D_MODEL, lane);
            wait_async_le16();
#pragma unroll
            for (int i = 0; i < 4; i++)   // warm next V tile
                __builtin_prefetch(Vt + (size_t)(w * 128 + i * 32 + lane) * NTOK
                                      + tokb + kv + 32, 0, 0);
        } else {
            wait_async_0();
        }

        // partial scores on this wave's 128-wide D slice (K from LDS)
        v8f s0 = zero8(), s1 = zero8();
#pragma unroll
        for (int c = 0; c < 4; c++) {
            int kd = c * 32;
            v16bf a  = load_a_bf16rows(Qs, QS_LD, ln, w * 128 + kd, half);
            v16bf b0 = load_b_contig(kcur + ln * KS_LD + kd + 16 * half);
            v16bf b1 = load_b_contig(kcur + (16 + ln) * KS_LD + kd + 16 * half);
            s0 = wmma_bf16(a, b0, s0);
            s1 = wmma_bf16(a, b1, s1);
        }

        __syncthreads();
        { float* scf = &sc[0][0]; scf[tid] = 0.f; scf[tid + 256] = 0.f; }
        __syncthreads();
#pragma unroll
        for (int v = 0; v < 8; v++) {     // cross-wave reduction (ds_add_f32)
            atomicAdd(&sc[v + 8 * half][ln],      s0[v]);
            atomicAdd(&sc[v + 8 * half][16 + ln], s1[v]);
        }
        __syncthreads();

        if (tid < 16) {                   // online softmax, one thread per row
            const int r = tid, rowg = q0 + r;
            float m = Mrow[r], mx = m;
#pragma unroll
            for (int c = 0; c < 32; c++) {
                float x = (kv + c > rowg) ? -3.0e38f : sc[r][c];
                mx = fmaxf(mx, x);
            }
            float alpha = __expf(m - mx);
            float l = Lrow[r] * alpha;
#pragma unroll
            for (int c = 0; c < 32; c++) {
                float p = (kv + c > rowg) ? 0.f : __expf(sc[r][c] - mx);
                l += p;
                Pt[r * PT_LD + c] = (bf16)p;
            }
            Mrow[r] = mx; Lrow[r] = l; Arow[r] = alpha;
        }
        __syncthreads();

        // rescale O and accumulate P * V (Vt rows contiguous in tokens)
        float av[8];
#pragma unroll
        for (int v = 0; v < 8; v++) av[v] = Arow[v + 8 * half];
        v16bf pa = load_a_bf16rows(Pt, PT_LD, ln, 0, half);
#pragma unroll
        for (int st = 0; st < 8; st++) {
#pragma unroll
            for (int v = 0; v < 8; v++) oacc[st][v] *= av[v];
            v16bf bv = load_b_contig(Vt + (size_t)(w * 128 + st * 16 + ln) * NTOK
                                        + tokb + kv + 16 * half);
            oacc[st] = wmma_bf16(pa, bv, oacc[st]);
        }
    }

    // ---- Epilogue: divide by softmax denominator, write fp32 output ----
#pragma unroll
    for (int v = 0; v < 8; v++) {
        int r = v + 8 * half;
        float rl = 1.f / Lrow[r];
#pragma unroll
        for (int st = 0; st < 8; st++)
            Out[(size_t)(tokb + q0 + r) * D_MODEL + w * 128 + st * 16 + ln] =
                oacc[st][v] * rl;
    }
}

// ---------------------------------------------------------------------------
extern "C" void kernel_launch(void* const* d_in, const int* in_sizes, int n_in,
                              void* d_out, int out_size, void* d_ws, size_t ws_size,
                              hipStream_t stream) {
    const float* X  = (const float*)d_in[0];
    const float* WK = (const float*)d_in[1];
    const float* WQ = (const float*)d_in[2];
    const float* WV = (const float*)d_in[3];

    const size_t NE = (size_t)NTOK * D_MODEL;        // 8M elems
    const size_t WE = (size_t)D_MODEL * D_MODEL;     // 1M elems
    bf16* Kbf = (bf16*)d_ws;
    bf16* Vt  = Kbf + NE;
    bf16* Xbf = Vt + NE;
    bf16* WtK = Xbf + NE;
    bf16* WtQ = WtK + WE;
    bf16* WtV = WtQ + WE;                            // total 54 MB workspace

    convert_x_kernel<<<dim3((unsigned)(NE / (256 * 8))), dim3(256), 0, stream>>>(X, Xbf);
    transpose_w_kernel<<<dim3(32, 32, 3), dim3(256), 0, stream>>>(WK, WQ, WV,
                                                                  WtK, WtQ, WtV);
    gemm_kv_kernel<<<dim3(NTOK / 16, D_MODEL / 64, 2), dim3(32), 0, stream>>>(
        Xbf, WtK, WtV, Kbf, Vt);
    attn_kernel<<<dim3(SEQ / 16, NBATCH), dim3(256), 0, stream>>>(
        Xbf, WtQ, Kbf, Vt, (float*)d_out);
}
